// MUSE_AttnProcessor_2800318677572
// MI455X (gfx1250) — compile-verified, matrix operand order checked by reading, not run on hardware
//
#include <hip/hip_runtime.h>
#include <hip/hip_bf16.h>
#include <stdint.h>

// ---------- types ----------
typedef __bf16 bf16t;
typedef bf16t v16bf __attribute__((ext_vector_type(16)));
typedef float v8f __attribute__((ext_vector_type(8)));
typedef unsigned int v4u __attribute__((ext_vector_type(4)));
typedef unsigned int u32x4_t __attribute__((ext_vector_type(4)));
typedef int i32x4_t __attribute__((ext_vector_type(4)));
typedef int i32x8_t __attribute__((ext_vector_type(8)));

union Frag {
    v16bf v;
    v4u   q[2];
    unsigned short us[16];
};

#ifndef __has_builtin
#define __has_builtin(x) 0
#endif
#if __has_builtin(__builtin_amdgcn_tensor_load_to_lds) && \
    __has_builtin(__builtin_amdgcn_s_wait_tensorcnt)
#define HAVE_TDM 1
#else
#define HAVE_TDM 0
#endif

// Raw split workgroup barrier: no implied vmem fence (unlike __syncthreads),
// so in-flight A-prefetch global loads stay outstanding across it.
#define WG_BARRIER() \
    asm volatile("s_barrier_signal -1\n\ts_barrier_wait -1" ::: "memory")

__device__ __forceinline__ unsigned short f2bf(float f) {
    unsigned int u = __float_as_uint(f);
    u += 0x7FFFu + ((u >> 16) & 1u);   // round-to-nearest-even
    return (unsigned short)(u >> 16);
}
__device__ __forceinline__ float bf2f(unsigned short h) {
    return __uint_as_float(((unsigned int)h) << 16);
}

#if HAVE_TDM
// Issue a TDM DMA of a 64-row x 64-halfword tile (row stride = K elements)
// from a global bf16 matrix into LDS at byte offset lds_addr.
// D# layout per CDNA5 ISA ch.8 (group0 128b, group1 256b; groups 2/3 zero).
__device__ __forceinline__ void tdm_load_tile_64x64(const unsigned short* gptr,
                                                    unsigned lds_addr, int K) {
    unsigned long long ga = (unsigned long long)(uintptr_t)gptr;
    u32x4_t g0;
    g0[0] = 1u;                                    // count=1, valid, no gather
    g0[1] = lds_addr;                              // lds_addr [63:32]
    g0[2] = (unsigned)(ga & 0xFFFFFFFFu);          // global_addr[31:0]
    g0[3] = (unsigned)((ga >> 32) & 0x01FFFFFFu)   // global_addr[56:32]
            | 0x80000000u;                         // type=2 ("image") [127:126]
    i32x8_t g1;
    g1[0] = 0x00010000;          // data_size=1 (2 bytes) at bits [17:16]
    g1[1] = (int)(64u << 16);    // tensor_dim0[15:0]=64 at bits [63:48]
    g1[2] = (int)(64u << 16);    // tensor_dim0 hi=0, tensor_dim1[15:0]=64
    g1[3] = (int)(64u << 16);    // tensor_dim1 hi=0, tile_dim0=64 [127:112]
    g1[4] = 64;                  // tile_dim1=64 [143:128], tile_dim2=0
    g1[5] = K;                   // tensor_dim0_stride[31:0] = K elements
    g1[6] = 0;                   // stride hi, tensor_dim1_stride lo
    g1[7] = 0;
    i32x4_t gz = {0, 0, 0, 0};
#if defined(__clang_major__) && __clang_major__ >= 23
    i32x8_t gz8 = {0, 0, 0, 0, 0, 0, 0, 0};
    __builtin_amdgcn_tensor_load_to_lds(g0, g1, gz, gz, gz8, 0);
#else
    __builtin_amdgcn_tensor_load_to_lds(g0, g1, gz, gz, 0);
#endif
}
#endif

// ---------- conversion kernels ----------
__global__ void k_f32_to_bf16(const float* __restrict__ in,
                              unsigned short* __restrict__ out, int n) {
    int i = blockIdx.x * blockDim.x + threadIdx.x;
    if (i < n) out[i] = f2bf(in[i]);
}

// Wt[n*K + k] = bf16(W[k*N + n])
__global__ void k_transpose_w(const float* __restrict__ W,
                              unsigned short* __restrict__ Wt, int K, int N) {
    int i = blockIdx.x * blockDim.x + threadIdx.x;
    if (i >= K * N) return;
    int n = i / K;
    int k = i - n * K;
    Wt[i] = f2bf(W[(size_t)k * N + n]);
}

// vout[(b*C + c)*Lpad + t] = vin[(b*Lpad + t)*C + c]  (zero padded tokens)
__global__ void k_transpose_v(const unsigned short* __restrict__ vin,
                              unsigned short* __restrict__ vout,
                              int C, int Lpad, int Lact, int total) {
    int i = blockIdx.x * blockDim.x + threadIdx.x;
    if (i >= total) return;
    int t = i % Lpad;
    int c = (i / Lpad) % C;
    int b = i / (Lpad * C);
    vout[i] = (t < Lact) ? vin[((size_t)(b * Lpad + t)) * C + c]
                         : (unsigned short)0;
}

// ---------- bf16 WMMA GEMM with TDM-staged B panel ----------
// C[M,N] = A[M,K] (bf16 row major) x Bt[N,K] (bf16 pre-transposed).
// Block: 256 threads = 8 waves, each wave owns a 16x64 output tile.
// Wave 0 DMAs the shared 64x64 B tile into LDS via tensor_load_to_lds,
// double-buffered; A fragments ping-pong in registers (loop unrolled by 2;
// K/64 is even for all K used here: 1280, 2048, 768).
// Output row mapping: m -> (m/rows_in)*rows_out + out_off + (m%rows_in).
// outf != null: f32 epilogue (+bias +residual), else bf16 store.
__global__ __launch_bounds__(256)
void k_gemm_bf16(const unsigned short* __restrict__ A,
                 const unsigned short* __restrict__ Bt,
                 int M, int K, int N, int rows_in, int rows_out, int out_off,
                 unsigned short* __restrict__ outb, float* __restrict__ outf,
                 const float* __restrict__ bias, const float* __restrict__ resid) {
    __shared__ __align__(16) unsigned short stage[2][64 * 64];
    const int tid   = threadIdx.x;
    const int lane  = tid & 31;
    const int half  = lane >> 4;
    const int mr    = lane & 15;
    const int waveu = __builtin_amdgcn_readfirstlane(tid >> 5); // wave-uniform
    const int n0    = blockIdx.x * 64;
    const int m0    = blockIdx.y * 128 + waveu * 16;

    int mA = m0 + mr;
    if (mA > M - 1) mA = M - 1;               // clamp ragged M for loads
    const unsigned short* arow = A + (size_t)mA * K;
    const int nchunk = K >> 6;                // 64-wide K chunks (even)

    v8f acc[4] = {{}, {}, {}, {}};

#if HAVE_TDM
    const unsigned short* bpanel = Bt + (size_t)n0 * K;
    unsigned lds0 = (unsigned)(uintptr_t)(&stage[0][0]);
    unsigned lds1 = (unsigned)(uintptr_t)(&stage[1][0]);

    Frag aE0, aE1, aO0, aO1;
    if (waveu == 0) tdm_load_tile_64x64(bpanel, lds0, K);
    aE0.q[0] = *(const v4u*)(arow + 8 * half);
    aE0.q[1] = *(const v4u*)(arow + 16 + 8 * half);
    aE1.q[0] = *(const v4u*)(arow + 32 + 8 * half);
    aE1.q[1] = *(const v4u*)(arow + 48 + 8 * half);

    auto kstep = [&](int i, int bufsel, Frag& c0, Frag& c1, Frag& p0, Frag& p1) {
        const bool more = (i + 1 < nchunk);
        if (more) {
            if (waveu == 0)
                tdm_load_tile_64x64(bpanel + (size_t)(i + 1) * 64,
                                    ((i + 1) & 1) ? lds1 : lds0, K);
            const int kb = (i + 1) << 6;
            p0.q[0] = *(const v4u*)(arow + kb + 8 * half);
            p0.q[1] = *(const v4u*)(arow + kb + 16 + 8 * half);
            p1.q[0] = *(const v4u*)(arow + kb + 32 + 8 * half);
            p1.q[1] = *(const v4u*)(arow + kb + 48 + 8 * half);
        }
        if (waveu == 0) {
            if (more) __builtin_amdgcn_s_wait_tensorcnt((short)1);
            else      __builtin_amdgcn_s_wait_tensorcnt((short)0);
        }
        WG_BARRIER();                       // chunk i resident in stage[bufsel]
        Frag bf0[4], bf1[4];
#pragma unroll
        for (int j = 0; j < 4; ++j) {       // one ds clause, one wait
            const unsigned short* bp = &stage[bufsel][(16 * j + mr) * 64];
            bf0[j].q[0] = *(const v4u*)(bp + 8 * half);
            bf0[j].q[1] = *(const v4u*)(bp + 16 + 8 * half);
            bf1[j].q[0] = *(const v4u*)(bp + 32 + 8 * half);
            bf1[j].q[1] = *(const v4u*)(bp + 48 + 8 * half);
        }
#pragma unroll
        for (int j = 0; j < 4; ++j) {
            acc[j] = __builtin_amdgcn_wmma_f32_16x16x32_bf16(
                false, c0.v, false, bf0[j].v, (short)0, acc[j], false, false);
            acc[j] = __builtin_amdgcn_wmma_f32_16x16x32_bf16(
                false, c1.v, false, bf1[j].v, (short)0, acc[j], false, false);
        }
        // b-fragment ds_loads must land before this buffer is re-DMA'd
        asm volatile("s_wait_dscnt 0x0" ::: "memory");
        WG_BARRIER();
    };

    for (int i = 0; i < nchunk; i += 2) {
        kstep(i,     0, aE0, aE1, aO0, aO1);
        kstep(i + 1, 1, aO0, aO1, aE0, aE1);
    }
#else
    // fallback: cooperative synchronous global->LDS copy, single buffer
    for (int i = 0; i < nchunk; ++i) {
        const int row = tid >> 2, cq = (tid & 3) * 16;
        v4u t0 = *(const v4u*)(Bt + (size_t)(n0 + row) * K + i * 64 + cq);
        v4u t1 = *(const v4u*)(Bt + (size_t)(n0 + row) * K + i * 64 + cq + 8);
        __syncthreads();
        *(v4u*)(&stage[0][row * 64 + cq]) = t0;
        *(v4u*)(&stage[0][row * 64 + cq + 8]) = t1;
        __syncthreads();
        Frag a0, a1;
        const int kb = i << 6;
        a0.q[0] = *(const v4u*)(arow + kb + 8 * half);
        a0.q[1] = *(const v4u*)(arow + kb + 16 + 8 * half);
        a1.q[0] = *(const v4u*)(arow + kb + 32 + 8 * half);
        a1.q[1] = *(const v4u*)(arow + kb + 48 + 8 * half);
        Frag bf0[4], bf1[4];
#pragma unroll
        for (int j = 0; j < 4; ++j) {
            const unsigned short* bp = &stage[0][(16 * j + mr) * 64];
            bf0[j].q[0] = *(const v4u*)(bp + 8 * half);
            bf0[j].q[1] = *(const v4u*)(bp + 16 + 8 * half);
            bf1[j].q[0] = *(const v4u*)(bp + 32 + 8 * half);
            bf1[j].q[1] = *(const v4u*)(bp + 48 + 8 * half);
        }
#pragma unroll
        for (int j = 0; j < 4; ++j) {
            acc[j] = __builtin_amdgcn_wmma_f32_16x16x32_bf16(
                false, a0.v, false, bf0[j].v, (short)0, acc[j], false, false);
            acc[j] = __builtin_amdgcn_wmma_f32_16x16x32_bf16(
                false, a1.v, false, bf1[j].v, (short)0, acc[j], false, false);
        }
        __syncthreads();
    }
#endif

#pragma unroll
    for (int j = 0; j < 4; ++j) {
#pragma unroll
        for (int r = 0; r < 8; ++r) {
            int m = m0 + r + 8 * half;
            if (m >= M) continue;
            int bb = m / rows_in;
            int rr = m - bb * rows_in;
            size_t orow = (size_t)bb * rows_out + out_off + rr;
            int col = n0 + 16 * j + mr;
            size_t idx = orow * (size_t)N + col;
            float val = acc[j][r];
            if (outf) {
                if (bias)  val += bias[col];
                if (resid) val += resid[idx];
                outf[idx] = val;
            } else {
                outb[idx] = f2bf(val);
            }
        }
    }
}

// ---------- attention ----------
// One block = 4 waves = 64 queries of one (batch, head). Context length <= 96.
// q: [B*S, C] bf16   k: [B, Lpad, C] bf16   vt: [B, C, Lpad] bf16
// out[idx] = bf16( P·V (+ addend) )   (addend fuses the two-branch sum)
__global__ __launch_bounds__(128)
void k_attn(const unsigned short* __restrict__ qb,
            const unsigned short* __restrict__ kb,
            const unsigned short* __restrict__ vt,
            const unsigned short* __restrict__ addend,
            unsigned short* __restrict__ outb,
            int Lact, int Lpad) {
    constexpr int S = 4096, C = 1280, H = 20;
    __shared__ __align__(16) float          sc[4][16][96];
    __shared__ __align__(16) unsigned short pb[4][16][96];

    const int w    = threadIdx.x >> 5;
    const int lane = threadIdx.x & 31;
    const int half = lane >> 4;
    const int mr   = lane & 15;

    const int qtiles = S / 64;
    int bid = blockIdx.x;
    int qt  = bid % qtiles;
    int h   = (bid / qtiles) % H;
    int b   = bid / (qtiles * H);
    const int q0 = qt * 64 + w * 16;

    const unsigned short* qrow = qb + ((size_t)(b * S + q0 + mr)) * C + h * 64;
    Frag qf0, qf1;
    qf0.q[0] = *(const v4u*)(qrow + 8 * half);
    qf0.q[1] = *(const v4u*)(qrow + 16 + 8 * half);
    qf1.q[0] = *(const v4u*)(qrow + 32 + 8 * half);
    qf1.q[1] = *(const v4u*)(qrow + 32 + 16 + 8 * half);

    // ---- scores = Q K^T, parked in LDS ----
    for (int t = 0; t < Lpad / 16; ++t) {
        const unsigned short* krow =
            kb + ((size_t)(b * Lpad + t * 16 + mr)) * C + h * 64;
        Frag kf0, kf1;
        kf0.q[0] = *(const v4u*)(krow + 8 * half);
        kf0.q[1] = *(const v4u*)(krow + 16 + 8 * half);
        kf1.q[0] = *(const v4u*)(krow + 32 + 8 * half);
        kf1.q[1] = *(const v4u*)(krow + 32 + 16 + 8 * half);
        v8f acc = {};
        acc = __builtin_amdgcn_wmma_f32_16x16x32_bf16(
            false, qf0.v, false, kf0.v, (short)0, acc, false, false);
        acc = __builtin_amdgcn_wmma_f32_16x16x32_bf16(
            false, qf1.v, false, kf1.v, (short)0, acc, false, false);
#pragma unroll
        for (int r = 0; r < 8; ++r)
            sc[w][r + 8 * half][t * 16 + mr] = acc[r];
    }
    __syncthreads();

    // ---- masked softmax (scale = d^-0.5 = 0.125), probs -> bf16 in LDS ----
    if (lane < 16) {
        const int row = lane;
        float mx = -3.0e38f;
        for (int j = 0; j < Lact; ++j) {
            float s = sc[w][row][j] * 0.125f;
            mx = s > mx ? s : mx;
        }
        float sum = 0.f;
        for (int j = 0; j < Lact; ++j) {
            float e = __expf(sc[w][row][j] * 0.125f - mx);
            sc[w][row][j] = e;
            sum += e;
        }
        float inv = 1.f / sum;
        for (int j = 0; j < Lpad; ++j)
            pb[w][row][j] = (j < Lact) ? f2bf(sc[w][row][j] * inv)
                                       : (unsigned short)0;
    }
    __syncthreads();

    // ---- out = P V ----
#pragma unroll
    for (int nt = 0; nt < 4; ++nt) {
        v8f acc = {};
        for (int t = 0; t < Lpad / 32; ++t) {
            Frag pf, vf;
            const unsigned short* prow = &pb[w][mr][t * 32];
            pf.q[0] = *(const v4u*)(prow + 8 * half);
            pf.q[1] = *(const v4u*)(prow + 16 + 8 * half);
            const unsigned short* vrow =
                vt + ((size_t)(b * C + h * 64 + nt * 16 + mr)) * Lpad + t * 32;
            vf.q[0] = *(const v4u*)(vrow + 8 * half);
            vf.q[1] = *(const v4u*)(vrow + 16 + 8 * half);
            acc = __builtin_amdgcn_wmma_f32_16x16x32_bf16(
                false, pf.v, false, vf.v, (short)0, acc, false, false);
        }
#pragma unroll
        for (int r = 0; r < 8; ++r) {
            int qr = q0 + r + 8 * half;
            size_t idx = ((size_t)(b * S + qr)) * C + h * 64 + nt * 16 + mr;
            float val = acc[r];
            if (addend) val += bf2f(addend[idx]);
            outb[idx] = f2bf(val);
        }
    }
}

// ---------- host launcher ----------
extern "C" void kernel_launch(void* const* d_in, const int* in_sizes, int n_in,
                              void* d_out, int out_size, void* d_ws, size_t ws_size,
                              hipStream_t stream) {
    (void)in_sizes; (void)n_in; (void)out_size; (void)ws_size;
    const float* hs   = (const float*)d_in[0];
    const float* ehs  = (const float*)d_in[1];
    const float* tbg  = (const float*)d_in[2];
    const float* ibg  = (const float*)d_in[3];
    const float* Wq   = (const float*)d_in[4];
    const float* Wk   = (const float*)d_in[5];
    const float* Wv   = (const float*)d_in[6];
    const float* Wkt  = (const float*)d_in[7];
    const float* Wvt  = (const float*)d_in[8];
    const float* Wki  = (const float*)d_in[9];
    const float* Wvi  = (const float*)d_in[10];
    const float* Wo   = (const float*)d_in[11];
    const float* bo   = (const float*)d_in[12];
    float* out = (float*)d_out;

    const int B = 4, S = 4096, C = 1280, L = 77, Dc = 2048, Dt = 768;
    const int Nt = 10, Ni = 40, H = 20;
    const int L1 = L + Nt, L1p = 96;    // text context 87 -> pad 96
    const int L2 = Ni,     L2p = 64;    // image context 40 -> pad 64
    const int Mq = B * S;               // 16384

    char* base = (char*)d_ws;
    size_t off = 0;
    auto alloc = [&](size_t bytes) -> unsigned short* {
        void* p = base + off;
        off = (off + bytes + 255) & ~(size_t)255;
        return (unsigned short*)p;
    };
    unsigned short* hs_bf    = alloc((size_t)Mq * C * 2);
    unsigned short* ehs_bf   = alloc((size_t)B * L * Dc * 2);
    unsigned short* tbg_bf   = alloc((size_t)B * Nt * Dt * 2);
    unsigned short* ibg_bf   = alloc((size_t)B * Ni * Dc * 2);
    unsigned short* Wq_t     = alloc((size_t)C * C * 2);
    unsigned short* Wk_t     = alloc((size_t)Dc * C * 2);
    unsigned short* Wv_t     = alloc((size_t)Dc * C * 2);
    unsigned short* Wkt_t    = alloc((size_t)Dt * C * 2);
    unsigned short* Wvt_t    = alloc((size_t)Dt * C * 2);
    unsigned short* Wki_t    = alloc((size_t)Dc * C * 2);
    unsigned short* Wvi_t    = alloc((size_t)Dc * C * 2);
    unsigned short* Wo_t     = alloc((size_t)C * C * 2);
    unsigned short* q_bf     = alloc((size_t)Mq * C * 2);
    unsigned short* k1_bf    = alloc((size_t)B * L1p * C * 2);
    unsigned short* v1_bf    = alloc((size_t)B * L1p * C * 2);
    unsigned short* v1t      = alloc((size_t)B * C * L1p * 2);
    unsigned short* k2_bf    = alloc((size_t)B * L2p * C * 2);
    unsigned short* v2_bf    = alloc((size_t)B * L2p * C * 2);
    unsigned short* v2t      = alloc((size_t)B * C * L2p * 2);
    unsigned short* attn1_bf = alloc((size_t)Mq * C * 2);
    unsigned short* hs2_bf   = alloc((size_t)Mq * C * 2);

    auto cvt = [&](const float* in, unsigned short* o, int n) {
        k_f32_to_bf16<<<(n + 255) / 256, 256, 0, stream>>>(in, o, n);
    };
    auto twt = [&](const float* W, unsigned short* Wt, int K, int N) {
        int n = K * N;
        k_transpose_w<<<(n + 255) / 256, 256, 0, stream>>>(W, Wt, K, N);
    };
    auto gemm = [&](const unsigned short* Aa, const unsigned short* Bb,
                    int M, int K, int N, int ri, int ro, int oo,
                    unsigned short* ob, float* of,
                    const float* bias, const float* resid) {
        dim3 g(N / 64, (M + 127) / 128);
        k_gemm_bf16<<<g, 256, 0, stream>>>(Aa, Bb, M, K, N, ri, ro, oo,
                                           ob, of, bias, resid);
    };

    // 1) convert activations, convert+transpose weights
    cvt(hs,  hs_bf,  Mq * C);
    cvt(ehs, ehs_bf, B * L * Dc);
    cvt(tbg, tbg_bf, B * Nt * Dt);
    cvt(ibg, ibg_bf, B * Ni * Dc);
    twt(Wq,  Wq_t,  C,  C);
    twt(Wk,  Wk_t,  Dc, C);
    twt(Wv,  Wv_t,  Dc, C);
    twt(Wkt, Wkt_t, Dt, C);
    twt(Wvt, Wvt_t, Dt, C);
    twt(Wki, Wki_t, Dc, C);
    twt(Wvi, Wvi_t, Dc, C);
    twt(Wo,  Wo_t,  C,  C);

    // 2) projections (TDM-staged WMMA GEMMs); k/v into padded [B,Lpad,C]
    gemm(hs_bf,  Wq_t,  Mq,     C,  C, Mq, Mq,  0, q_bf,  nullptr, nullptr, nullptr);
    gemm(ehs_bf, Wk_t,  B * L,  Dc, C, L,  L1p, 0, k1_bf, nullptr, nullptr, nullptr);
    gemm(ehs_bf, Wv_t,  B * L,  Dc, C, L,  L1p, 0, v1_bf, nullptr, nullptr, nullptr);
    gemm(tbg_bf, Wkt_t, B * Nt, Dt, C, Nt, L1p, L, k1_bf, nullptr, nullptr, nullptr);
    gemm(tbg_bf, Wvt_t, B * Nt, Dt, C, Nt, L1p, L, v1_bf, nullptr, nullptr, nullptr);
    gemm(ibg_bf, Wki_t, B * Ni, Dc, C, Ni, L2p, 0, k2_bf, nullptr, nullptr, nullptr);
    gemm(ibg_bf, Wvi_t, B * Ni, Dc, C, Ni, L2p, 0, v2_bf, nullptr, nullptr, nullptr);

    // 3) transpose V for PV fragment loads (zero padded tokens)
    int tv1 = B * C * L1p;
    k_transpose_v<<<(tv1 + 255) / 256, 256, 0, stream>>>(v1_bf, v1t, C, L1p, L1, tv1);
    int tv2 = B * C * L2p;
    k_transpose_v<<<(tv2 + 255) / 256, 256, 0, stream>>>(v2_bf, v2t, C, L2p, L2, tv2);

    // 4) attention: branch 1 (text), then branch 2 (image) fusing the sum
    int nblk = B * H * (S / 64);
    k_attn<<<nblk, 128, 0, stream>>>(q_bf, k1_bf, v1t, nullptr,  attn1_bf, L1, L1p);
    k_attn<<<nblk, 128, 0, stream>>>(q_bf, k2_bf, v2t, attn1_bf, hs2_bf,   L2, L2p);

    // 5) output projection + bias + residual -> f32 d_out
    gemm(hs2_bf, Wo_t, Mq, C, C, Mq, Mq, 0, nullptr, out, bo, hs);
}
